// CrossMultiHeadAttention_12867722018932
// MI455X (gfx1250) — compile-verified, matrix operand order checked by reading, not run on hardware
//
#include <hip/hip_runtime.h>
#include <hip/hip_bf16.h>

// ---------------------------------------------------------------------------
// CDNA5 (gfx1250) cross multi-head attention, wave32 + V_WMMA_F32_16X16X32_F16
// ---------------------------------------------------------------------------

typedef __attribute__((ext_vector_type(16))) _Float16 v16h;
typedef __attribute__((ext_vector_type(8)))  float    v8f;

#define NUM_HEADS 16
#define HEAD_DIM  64
#define DMODEL    1024
#define SEQ_N     2048
#define SEQ_M     2048
#define BATCH     4

union AFrag { v16h v; _Float16 h[16]; };

// Load a 16x32 f16 A/B fragment from an LDS tile stored row-major over K.
// ISA 16-bit A layout: lane m=lane&15 (row), half=lane>>4;
// VGPR v<4  packs K = 2v   + 8*half, +1
// VGPR v>=4 packs K = 16+2(v-4) + 8*half, +1
__device__ __forceinline__ v16h ldsFragRow(const _Float16* base, int row,
                                           int stride, int lane) {
  const int m = lane & 15;
  const int lh = lane >> 4;
  const _Float16* rp = base + (size_t)(row + m) * stride;
  AFrag f;
#pragma unroll
  for (int v = 0; v < 8; ++v) {
    int k = (v < 4) ? (2 * v + 8 * lh) : (16 + 2 * (v - 4) + 8 * lh);
    f.h[2 * v]     = rp[k];
    f.h[2 * v + 1] = rp[k + 1];
  }
  return f.v;
}

__device__ __forceinline__ v8f wmma32(v16h a, v16h b, v8f c) {
  return __builtin_amdgcn_wmma_f32_16x16x32_f16(false, a, false, b,
                                                (short)0, c, false, false);
}

// ---------------------------------------------------------------------------
// C[rows x 1024] = A[rows x 1024] * W^T   (W is [1024 out][1024 in] row-major)
// Block = 256 threads (8 waves). Tile 128x64, BK=32, double-buffered LDS with
// register staging: one barrier per k-step, 4 WMMAs per wave per k-step.
// SCATTER: write f16 head-major [B, H, rowsPerBatch, 64]; else f32 row-major.
// ---------------------------------------------------------------------------
template <typename AT, bool SCATTER>
__global__ __launch_bounds__(256) void gemm_xwT(const AT* __restrict__ A,
                                                const float* __restrict__ W,
                                                void* __restrict__ outp,
                                                int rowsPerBatch) {
  __shared__ _Float16 sA[2][128][40];
  __shared__ _Float16 sB[2][64][40];

  const int tid  = threadIdx.x;
  const int lane = tid & 31;
  const int wave = tid >> 5;
  const int row0 = blockIdx.y * 128;
  const int col0 = blockIdx.x * 64;
  const int wr = (wave >> 1) * 32;   // wave row offset in tile (0,32,64,96)
  const int wc = (wave & 1) * 32;    // wave col offset in tile (0,32)

  // LDS fill mapping: A tile 128x32 -> 16 elems/thread; B tile 64x32 -> 8.
  const int ar = tid >> 1;           // 0..127
  const int ac = (tid & 1) * 16;     // 0,16
  const int br = tid >> 2;           // 0..63
  const int bc = (tid & 3) * 8;      // 0,8,16,24

  float ra[16];
  float rb[8];

  // Preload k-tile 0 into registers, stage to LDS buffer 0.
  {
    const AT*    ap = A + (size_t)(row0 + ar) * DMODEL + ac;
    const float* wp = W + (size_t)(col0 + br) * DMODEL + bc;
#pragma unroll
    for (int u = 0; u < 16; ++u) ra[u] = (float)ap[u];
#pragma unroll
    for (int u = 0; u < 8; ++u) rb[u] = wp[u];
#pragma unroll
    for (int u = 0; u < 16; ++u) sA[0][ar][ac + u] = (_Float16)ra[u];
#pragma unroll
    for (int u = 0; u < 8; ++u) sB[0][br][bc + u] = (_Float16)rb[u];
  }
  __syncthreads();

  v8f o00 = {}, o01 = {}, o10 = {}, o11 = {};

  constexpr int NSTEP = DMODEL / 32;  // 32
  for (int it = 0; it < NSTEP; ++it) {
    const int cur = it & 1;
    const bool haveNext = (it + 1) < NSTEP;

    if (haveNext) {  // global loads for tile k+1 overlap the WMMAs below
      const int k0 = (it + 1) * 32;
      const AT*    ap = A + (size_t)(row0 + ar) * DMODEL + k0 + ac;
      const float* wp = W + (size_t)(col0 + br) * DMODEL + k0 + bc;
#pragma unroll
      for (int u = 0; u < 16; ++u) ra[u] = (float)ap[u];
#pragma unroll
      for (int u = 0; u < 8; ++u) rb[u] = wp[u];
      if (it + 2 < NSTEP) {
        __builtin_prefetch(ap + 32, 0, 1);   // global_prefetch_b8
        __builtin_prefetch(wp + 32, 0, 1);
      }
    }

    v16h a0 = ldsFragRow(&sA[cur][0][0], wr,      40, lane);
    v16h a1 = ldsFragRow(&sA[cur][0][0], wr + 16, 40, lane);
    v16h b0 = ldsFragRow(&sB[cur][0][0], wc,      40, lane);
    v16h b1 = ldsFragRow(&sB[cur][0][0], wc + 16, 40, lane);
    o00 = wmma32(a0, b0, o00);
    o01 = wmma32(a0, b1, o01);
    o10 = wmma32(a1, b0, o10);
    o11 = wmma32(a1, b1, o11);

    if (haveNext) {  // stage into the other buffer; no reader conflicts
      const int nxt = cur ^ 1;
#pragma unroll
      for (int u = 0; u < 16; ++u) sA[nxt][ar][ac + u] = (_Float16)ra[u];
#pragma unroll
      for (int u = 0; u < 8; ++u) sB[nxt][br][bc + u] = (_Float16)rb[u];
    }
    __syncthreads();
  }

  // C layout: VGPR e -> row = e + 8*(lane>=16); col = lane&15.
  const int lh = lane >> 4;
  const int nl = lane & 15;
#pragma unroll
  for (int sr = 0; sr < 2; ++sr) {
#pragma unroll
    for (int sc = 0; sc < 2; ++sc) {
      const v8f& o = (sr == 0) ? (sc == 0 ? o00 : o01) : (sc == 0 ? o10 : o11);
#pragma unroll
      for (int e = 0; e < 8; ++e) {
        const int gr = row0 + wr + sr * 16 + e + 8 * lh;
        const int gc = col0 + wc + sc * 16 + nl;
        if (SCATTER) {
          _Float16* out = (_Float16*)outp;
          const int b = gr / rowsPerBatch;
          const int n = gr % rowsPerBatch;
          const int h = gc >> 6, d = gc & 63;
          out[(((size_t)b * NUM_HEADS + h) * rowsPerBatch + n) * HEAD_DIM + d] =
              (_Float16)o[e];
        } else {
          ((float*)outp)[(size_t)gr * DMODEL + gc] = o[e];
        }
      }
    }
  }
}

// ---------------------------------------------------------------------------
// Flash attention: one block = one (b,h) and a 64-row query tile.
// S = Q K^T / 8, online softmax, O += P V, all tiles in LDS, WMMA throughout.
// V is stored TRANSPOSED in LDS (sVt[d][m]) so the P*V B-fragments are
// contiguous row-major reads. Softmax runs 4 threads per row; partial max/sum
// are combined with __shfl_xor over lane masks 1,2 (within an aligned quad).
// ---------------------------------------------------------------------------
__global__ __launch_bounds__(256) void flash_attn(
    const _Float16* __restrict__ Q, const _Float16* __restrict__ K,
    const _Float16* __restrict__ V, _Float16* __restrict__ ctx) {
  __shared__ _Float16 sQ[64][72];
  __shared__ _Float16 sK[64][72];
  __shared__ _Float16 sVt[64][72];   // [d][m] (transposed V tile)
  __shared__ _Float16 sP[64][72];
  __shared__ float    sS[64][68];
  __shared__ float    sAlpha[64];
  __shared__ float    sL[64];

  const int tid  = threadIdx.x;
  const int lane = tid & 31;
  const int wave = tid >> 5;
  const int n0 = blockIdx.x * 64;
  const int h  = blockIdx.y;
  const int b  = blockIdx.z;

  const _Float16* Qb = Q + (((size_t)b * NUM_HEADS + h) * SEQ_N + n0) * HEAD_DIM;
  const _Float16* Kb = K + ((size_t)b * NUM_HEADS + h) * SEQ_M * HEAD_DIM;
  const _Float16* Vb = V + ((size_t)b * NUM_HEADS + h) * SEQ_M * HEAD_DIM;

  const int lr = tid >> 2;
  const int lc = (tid & 3) * 16;
#pragma unroll
  for (int u = 0; u < 16; ++u) sQ[lr][lc + u] = Qb[(size_t)lr * HEAD_DIM + lc + u];

  const int wr = (wave >> 1) * 16;
  const int wc = (wave & 1) * 32;
  const int lh = lane >> 4;
  const int nl = lane & 15;

  // Softmax mapping: 4 threads per row, 16 columns each. All four lanes of a
  // quad carry identical (m,l) state; they reduce via shfl_xor 1,2.
  const int srow = tid >> 2;        // 0..63
  const int scol = (tid & 3) * 16;  // 0,16,32,48

  float m_i = -3.0e38f;
  float l_i = 0.0f;
  v8f o0 = {};
  v8f o1 = {};

  for (int m0 = 0; m0 < SEQ_M; m0 += 64) {
    __syncthreads();   // all waves done reading sK/sVt/sP of previous tile
#pragma unroll
    for (int u = 0; u < 16; ++u)
      sK[lr][lc + u] = Kb[(size_t)(m0 + lr) * HEAD_DIM + lc + u];
#pragma unroll
    for (int u = 0; u < 16; ++u)   // transpose V into LDS: sVt[d][m]
      sVt[lc + u][lr] = Vb[(size_t)(m0 + lr) * HEAD_DIM + lc + u];
    __syncthreads();

    // ---- S = Q K^T (per wave: 16x32 of the 64x64 tile), K-dim 64 = 2 steps
    v16h aq0 = ldsFragRow(&sQ[0][0],      wr, 72, lane);
    v16h aq1 = ldsFragRow(&sQ[0][0] + 32, wr, 72, lane);
    v8f s0 = {};
    v8f s1 = {};
    {
      v16h bk = ldsFragRow(&sK[0][0],      wc,      72, lane);
      s0 = wmma32(aq0, bk, s0);
      bk      = ldsFragRow(&sK[0][0] + 32, wc,      72, lane);
      s0 = wmma32(aq1, bk, s0);
      bk      = ldsFragRow(&sK[0][0],      wc + 16, 72, lane);
      s1 = wmma32(aq0, bk, s1);
      bk      = ldsFragRow(&sK[0][0] + 32, wc + 16, 72, lane);
      s1 = wmma32(aq1, bk, s1);
    }
#pragma unroll
    for (int e = 0; e < 8; ++e) {
      const int r = wr + e + 8 * lh;
      sS[r][wc + nl]      = s0[e] * 0.125f;   // 1/sqrt(64)
      sS[r][wc + 16 + nl] = s1[e] * 0.125f;
    }
    __syncthreads();

    // ---- online softmax: 4 lanes per row, quad reduction via shfl_xor
    {
      float mt = -3.0e38f;
#pragma unroll
      for (int j = 0; j < 16; ++j) mt = fmaxf(mt, sS[srow][scol + j]);
      mt = fmaxf(mt, __shfl_xor(mt, 1));
      mt = fmaxf(mt, __shfl_xor(mt, 2));
      const float mn = fmaxf(m_i, mt);
      const float alpha = __expf(m_i - mn);
      float rs = 0.0f;
#pragma unroll
      for (int j = 0; j < 16; ++j) {
        const float p = __expf(sS[srow][scol + j] - mn);
        sP[srow][scol + j] = (_Float16)p;
        rs += p;
      }
      rs += __shfl_xor(rs, 1);
      rs += __shfl_xor(rs, 2);
      l_i = l_i * alpha + rs;
      m_i = mn;
      if ((tid & 3) == 0) sAlpha[srow] = alpha;
    }
    __syncthreads();

    // ---- rescale O, then O += P V (P: 16x64 f16, V^T rows are O columns)
#pragma unroll
    for (int e = 0; e < 8; ++e) {
      const float a = sAlpha[wr + e + 8 * lh];
      o0[e] *= a;
      o1[e] *= a;
    }
    v16h ap0 = ldsFragRow(&sP[0][0],      wr, 72, lane);
    v16h ap1 = ldsFragRow(&sP[0][0] + 32, wr, 72, lane);
    {
      v16h bv = ldsFragRow(&sVt[0][0],      wc,      72, lane);
      o0 = wmma32(ap0, bv, o0);
      bv      = ldsFragRow(&sVt[0][0] + 32, wc,      72, lane);
      o0 = wmma32(ap1, bv, o0);
      bv      = ldsFragRow(&sVt[0][0],      wc + 16, 72, lane);
      o1 = wmma32(ap0, bv, o1);
      bv      = ldsFragRow(&sVt[0][0] + 32, wc + 16, 72, lane);
      o1 = wmma32(ap1, bv, o1);
    }
  }

  __syncthreads();
  if ((tid & 3) == 0) sL[srow] = l_i;
  __syncthreads();

#pragma unroll
  for (int e = 0; e < 8; ++e) {
    const int r = wr + e + 8 * lh;
    const float inv = 1.0f / sL[r];
    // ctx layout [B, N, D] with D = h*64 + d (ready for the output projection)
    _Float16* cp = ctx + ((size_t)b * SEQ_N + n0 + r) * DMODEL + h * HEAD_DIM;
    cp[wc + nl]      = (_Float16)(o0[e] * inv);
    cp[wc + 16 + nl] = (_Float16)(o1[e] * inv);
  }
}

// ---------------------------------------------------------------------------
extern "C" void kernel_launch(void* const* d_in, const int* in_sizes, int n_in,
                              void* d_out, int out_size, void* d_ws,
                              size_t ws_size, hipStream_t stream) {
  const float* inputs = (const float*)d_in[0];  // [B,N,D]
  const float* encOut = (const float*)d_in[1];  // [B,M,D]
  const float* wq = (const float*)d_in[2];
  const float* wk = (const float*)d_in[3];
  const float* wv = (const float*)d_in[4];
  const float* wo = (const float*)d_in[5];
  float* out = (float*)d_out;

  const size_t elems = (size_t)BATCH * SEQ_N * DMODEL;  // 8,388,608
  _Float16* qbuf = (_Float16*)d_ws;                     // [B,H,N,64]
  _Float16* kbuf = qbuf + elems;                        // [B,H,M,64]
  _Float16* vbuf = kbuf + elems;                        // [B,H,M,64]
  _Float16* cbuf = vbuf + elems;                        // [B,N,D]

  dim3 blk(256);
  dim3 gProj(DMODEL / 64, (BATCH * SEQ_N) / 128);       // 16 x 64

  gemm_xwT<float, true><<<gProj, blk, 0, stream>>>(inputs, wq, qbuf, SEQ_N);
  gemm_xwT<float, true><<<gProj, blk, 0, stream>>>(encOut, wk, kbuf, SEQ_M);
  gemm_xwT<float, true><<<gProj, blk, 0, stream>>>(encOut, wv, vbuf, SEQ_M);

  dim3 gAttn(SEQ_N / 64, NUM_HEADS, BATCH);             // 32 x 16 x 4
  flash_attn<<<gAttn, blk, 0, stream>>>(qbuf, kbuf, vbuf, cbuf);

  gemm_xwT<_Float16, false><<<gProj, blk, 0, stream>>>(cbuf, wo, out, SEQ_N);
}
